// attention_73375221285320
// MI455X (gfx1250) — matrix operand order repaired and checked by reading, not compiled, not run on hardware
//
#include <hip/hip_runtime.h>
#include <hip/hip_bf16.h>

typedef __attribute__((ext_vector_type(16))) __bf16 v16bf;
typedef __attribute__((ext_vector_type(8)))  float  v8f;
typedef __attribute__((ext_vector_type(2)))  float  v2f;
typedef __attribute__((ext_vector_type(8)))  unsigned int v8u;

#define BATCH 4
#define CH    64
#define C8    8
#define HW    64
#define NPIX  4096   // 64*64

// ---- CDNA5 async global->LDS path (guarded; falls back to plain staging) ----
#if defined(__HIP_DEVICE_COMPILE__) && \
    __has_builtin(__builtin_amdgcn_global_load_async_to_lds_b32) && \
    __has_builtin(__builtin_amdgcn_s_wait_asynccnt)
#define USE_ASYNC 1
#else
#define USE_ASYNC 0
#endif

#if USE_ASYNC
typedef __attribute__((address_space(1))) int as1_int;
typedef __attribute__((address_space(3))) int as3_int;
#endif

static __device__ __forceinline__ void async_g2l_b32(const float* g, float* l) {
#if USE_ASYNC
    __builtin_amdgcn_global_load_async_to_lds_b32(
        (as1_int*)(float*)g,   // addrspacecast flat->global, param is AS1 int*
        (as3_int*)l,           // addrspacecast flat->lds,    param is AS3 int*
        0, 0);
#else
    *l = *g;   // plain staging fallback (DS ops are wave-in-order)
#endif
}

#if USE_ASYNC
#define WAIT_ASYNC(n) __builtin_amdgcn_s_wait_asynccnt(n)
#else
#define WAIT_ASYNC(n) ((void)0)
#endif

static __device__ __forceinline__ unsigned short f32_to_bf16_bits(float f) {
    unsigned u = __float_as_uint(f);
    unsigned r = (u + 0x7FFFu + ((u >> 16) & 1u)) >> 16;   // round-to-nearest-even
    return (unsigned short)r;
}

// ---------------------------------------------------------------------------
// Pack OIHW fp32 3x3 weights into bf16 B-fragment order:
//   dst[(((tap*2+kc)*64 + o)*2 + kh)*16 + j]  with cin = kc*32 + kh*16 + j
// ---------------------------------------------------------------------------
__global__ __launch_bounds__(256) void pack_w_kernel(const float* __restrict__ src,
                                                     unsigned short* __restrict__ dst) {
    int t = blockIdx.x * 256 + threadIdx.x;
    if (t >= 9 * 2 * 64 * 2 * 16) return;
    int j   = t & 15;
    int kh  = (t >> 4) & 1;
    int o   = (t >> 5) & 63;
    int kc  = (t >> 11) & 1;
    int tap = t >> 12;
    int cin = kc * 32 + kh * 16 + j;
    dst[t] = f32_to_bf16_bits(src[(o * 64 + cin) * 9 + tap]);
}

// ---------------------------------------------------------------------------
// 3x3 conv, pad 1, 64->64 ch, implicit GEMM with v_wmma_f32_16x16x32_bf16.
// One block = one (batch, output row); 16 waves, each owning a 16x16 tile.
// ---------------------------------------------------------------------------
__global__ __launch_bounds__(512) void conv3x3_wmma_kernel(const float* __restrict__ in,
                                                           const unsigned short* __restrict__ pw,
                                                           float* __restrict__ out) {
    __shared__ __align__(16) unsigned short lds[3 * 66 * 64];

    const int b = blockIdx.x >> 6;
    const int y = blockIdx.x & 63;

    for (int idx = threadIdx.x; idx < 3 * 66 * 64; idx += 512) {
        int dy  = idx / (66 * 64);
        int rem = idx - dy * (66 * 64);
        int xp  = rem >> 6;
        int cin = rem & 63;
        int yy = y + dy - 1;
        int xx = xp - 1;
        float v = 0.0f;
        if ((unsigned)yy < 64u && (unsigned)xx < 64u)
            v = in[((b * CH + cin) * HW + yy) * HW + xx];
        lds[idx] = f32_to_bf16_bits(v);
    }
    __syncthreads();

    const int lane = threadIdx.x & 31;
    const int wave = threadIdx.x >> 5;          // 0..15
    const int mt = wave & 3;                    // spatial-x tile
    const int nt = wave >> 2;                   // outch tile
    const int mi = lane & 15;
    const int kh = lane >> 4;

    v8f acc = {0.f, 0.f, 0.f, 0.f, 0.f, 0.f, 0.f, 0.f};

#pragma unroll
    for (int tap = 0; tap < 9; ++tap) {
        const int dy = tap / 3;
        const int dx = tap % 3;
        const int xp = mt * 16 + mi + dx;       // 0..65 (halo-shifted index)
        if (tap < 8) {                          // prefetch next tap's weight line
            __builtin_prefetch(&pw[(((tap + 1) * 2 * 64) + nt * 16 + mi) * 32], 0, 1);
        }
#pragma unroll
        for (int kc = 0; kc < 2; ++kc) {
            const int abase = (dy * 66 + xp) * 64 + kc * 32 + kh * 16;
            uint4 q0 = *(const uint4*)&lds[abase];
            uint4 q1 = *(const uint4*)&lds[abase + 8];
            v8u au; au[0]=q0.x; au[1]=q0.y; au[2]=q0.z; au[3]=q0.w;
                    au[4]=q1.x; au[5]=q1.y; au[6]=q1.z; au[7]=q1.w;
            const unsigned short* wp =
                &pw[(((tap * 2 + kc) * 64) + nt * 16 + mi) * 32 + kh * 16];
            uint4 w0 = *(const uint4*)wp;
            uint4 w1 = *(const uint4*)(wp + 8);
            v8u bu; bu[0]=w0.x; bu[1]=w0.y; bu[2]=w0.z; bu[3]=w0.w;
                    bu[4]=w1.x; bu[5]=w1.y; bu[6]=w1.z; bu[7]=w1.w;

            v16bf av = __builtin_bit_cast(v16bf, au);
            v16bf bv = __builtin_bit_cast(v16bf, bu);
            acc = __builtin_amdgcn_wmma_f32_16x16x32_bf16(
                      false, av, false, bv, (short)0, acc, false, false);
        }
    }

    const int x0 = mt * 16;
    const int oc = nt * 16 + mi;
#pragma unroll
    for (int r = 0; r < 8; ++r) {
        int mm = r + 8 * kh;
        out[((b * CH + oc) * HW + y) * HW + x0 + mm] = acc[r];
    }
}

// ---------------------------------------------------------------------------
// f/g/h 1x1 projections + bias + folded BN (+gamma for h).
// ---------------------------------------------------------------------------
__global__ __launch_bounds__(256) void proj_kernel(
    const float* __restrict__ out1,
    const float* __restrict__ f_w, const float* __restrict__ f_b,
    const float* __restrict__ f_bw, const float* __restrict__ f_bb,
    const float* __restrict__ f_bm, const float* __restrict__ f_bv,
    const float* __restrict__ g_w, const float* __restrict__ g_b,
    const float* __restrict__ g_bw, const float* __restrict__ g_bb,
    const float* __restrict__ g_bm, const float* __restrict__ g_bv,
    const float* __restrict__ gamma,
    const float* __restrict__ h_w, const float* __restrict__ h_b,
    const float* __restrict__ h_bw, const float* __restrict__ h_bb,
    const float* __restrict__ h_bm, const float* __restrict__ h_bv,
    float* __restrict__ fbuf, float* __restrict__ gbuf, float* __restrict__ hbuf) {
    int t = blockIdx.x * 256 + threadIdx.x;
    int b = t >> 12;
    int n = t & 4095;

    float af[C8], ag[C8], ah[C8];
#pragma unroll
    for (int c = 0; c < C8; ++c) { af[c] = 0.f; ag[c] = 0.f; ah[c] = 0.f; }

    for (int cm = 0; cm < CH; ++cm) {
        float xv = out1[(b * CH + cm) * NPIX + n];
#pragma unroll
        for (int c = 0; c < C8; ++c) {
            af[c] = fmaf(f_w[c * CH + cm], xv, af[c]);
            ag[c] = fmaf(g_w[c * CH + cm], xv, ag[c]);
            ah[c] = fmaf(h_w[c * CH + cm], xv, ah[c]);
        }
    }
    float gm = gamma[0];
#pragma unroll
    for (int c = 0; c < C8; ++c) {
        float fs = f_bw[c] * rsqrtf(f_bv[c] + 1e-5f);
        float fo = f_bb[c] - f_bm[c] * fs;
        fbuf[(b * C8 + c) * NPIX + n] = (af[c] + f_b[c]) * fs + fo;
        float gs = g_bw[c] * rsqrtf(g_bv[c] + 1e-5f);
        float go = g_bb[c] - g_bm[c] * gs;
        gbuf[(b * C8 + c) * NPIX + n] = (ag[c] + g_b[c]) * gs + go;
        float hs = h_bw[c] * rsqrtf(h_bv[c] + 1e-5f);
        float ho = h_bb[c] - h_bm[c] * hs;
        hbuf[(b * C8 + c) * NPIX + n] = gm * ((ah[c] + h_b[c]) * hs + ho);
    }
}

// ---------------------------------------------------------------------------
// Pass 1: online column stats of S = f^T g. Per-iteration A-tile (512B) is
// double-buffered into LDS via async global->LDS DMA; B-frag hoisted.
// ---------------------------------------------------------------------------
__global__ __launch_bounds__(128) void att_stats_kernel(const float* __restrict__ f,
                                                        const float* __restrict__ g,
                                                        float* __restrict__ colmax,
                                                        float* __restrict__ colsum) {
    __shared__ __align__(16) float ldsA[4][2][128];

    const int wave = threadIdx.x >> 5;
    const int lane = threadIdx.x & 31;
    const int mi = lane & 15;
    const int kh = lane >> 4;
    const int b  = blockIdx.x >> 6;
    const int mb = (blockIdx.x & 63) * 64 + wave * 16;

    const float* fb = f + (size_t)b * C8 * NPIX;
    const float* gb = g + (size_t)b * C8 * NPIX;

    v2f b0, b1;
    b0[0] = gb[(2 * kh + 0) * NPIX + mb + mi];
    b0[1] = gb[(2 * kh + 1) * NPIX + mb + mi];
    b1[0] = gb[(4 + 2 * kh + 0) * NPIX + mb + mi];
    b1[1] = gb[(4 + 2 * kh + 1) * NPIX + mb + mi];

    auto issueA = [&](int nt, int buf) {
#pragma unroll
        for (int i = 0; i < 4; ++i) {
            int e = i * 32 + lane;               // e = c*16 + col
            async_g2l_b32(fb + (e >> 4) * NPIX + nt * 16 + (e & 15),
                          &ldsA[wave][buf][e]);
        }
    };

    issueA(0, 0);

    float runmax = -1e30f, runsum = 0.0f;

    for (int nt = 0; nt < 256; ++nt) {
        const int buf = nt & 1;
        if (nt + 1 < 256) {
            issueA(nt + 1, buf ^ 1);
            WAIT_ASYNC(4);      // in-order completion => current tile landed
        } else {
            WAIT_ASYNC(0);
        }
        const float* A = ldsA[wave][buf];
        v2f a0, a1;
        a0[0] = A[(2 * kh + 0) * 16 + mi];
        a0[1] = A[(2 * kh + 1) * 16 + mi];
        a1[0] = A[(4 + 2 * kh + 0) * 16 + mi];
        a1[1] = A[(4 + 2 * kh + 1) * 16 + mi];

        v8f s = {0.f, 0.f, 0.f, 0.f, 0.f, 0.f, 0.f, 0.f};
        s = __builtin_amdgcn_wmma_f32_16x16x4_f32(false, a0, false, b0, (short)0, s, false, false);
        s = __builtin_amdgcn_wmma_f32_16x16x4_f32(false, a1, false, b1, (short)0, s, false, false);

        float tmax = s[0];
#pragma unroll
        for (int r = 1; r < 8; ++r) tmax = fmaxf(tmax, s[r]);
        if (tmax > runmax) {
            runsum *= __expf(runmax - tmax);
            runmax = tmax;
        }
#pragma unroll
        for (int r = 0; r < 8; ++r) runsum += __expf(s[r] - runmax);
    }

    float om = __shfl_xor(runmax, 16, 32);
    float os = __shfl_xor(runsum, 16, 32);
    float nm = fmaxf(runmax, om);
    float ns = runsum * __expf(runmax - nm) + os * __expf(om - nm);
    if (lane < 16) {
        colmax[b * NPIX + mb + lane] = nm;
        colsum[b * NPIX + mb + lane] = ns;
    }
}

// Fold h[c][m] /= Z[m] so pass 2 is a plain weighted sum.
__global__ __launch_bounds__(256) void fold_h_kernel(float* __restrict__ hbuf,
                                                     const float* __restrict__ colsum) {
    int t = blockIdx.x * 256 + threadIdx.x;      // over B*8*4096
    int m = t & 4095;
    int b = t >> 15;
    hbuf[t] = hbuf[t] * (1.0f / colsum[b * NPIX + m]);
}

// ---------------------------------------------------------------------------
// Pass 2: h_att[n,c] = sum_m exp(S[n,m]-M[m]) * (h[m,c]/Z[m]).
// g- and h-tiles double-buffered via async global->LDS (8 ops/group);
// score-tile D->A re-layout through a 1KB per-wave LDS bounce.
// ---------------------------------------------------------------------------
__global__ __launch_bounds__(128) void att_apply_kernel(const float* __restrict__ f,
                                                        const float* __restrict__ g,
                                                        const float* __restrict__ h,
                                                        const float* __restrict__ colmax,
                                                        float* __restrict__ hatt) {
    __shared__ __align__(16) float etile[4][256];
    __shared__ __align__(16) float ldsGH[4][2][256];   // [0..127]=g tile, [128..255]=h tile

    const int wave = threadIdx.x >> 5;
    const int lane = threadIdx.x & 31;
    const int mi = lane & 15;
    const int kh = lane >> 4;
    const int b  = blockIdx.x >> 6;
    const int nb = (blockIdx.x & 63) * 64 + wave * 16;

    const float* fb = f + (size_t)b * C8 * NPIX;
    const float* gb = g + (size_t)b * C8 * NPIX;
    const float* hb = h + (size_t)b * C8 * NPIX;

    v2f a0, a1;   // rows fixed -> hoist A
    a0[0] = fb[(2 * kh + 0) * NPIX + nb + mi];
    a0[1] = fb[(2 * kh + 1) * NPIX + nb + mi];
    a1[0] = fb[(4 + 2 * kh + 0) * NPIX + nb + mi];
    a1[1] = fb[(4 + 2 * kh + 1) * NPIX + nb + mi];

    auto issueGH = [&](int mt, int buf) {
#pragma unroll
        for (int i = 0; i < 4; ++i) {
            int e = i * 32 + lane;               // e = c*16 + col
            int c = e >> 4, col = e & 15;
            async_g2l_b32(gb + c * NPIX + mt * 16 + col, &ldsGH[wave][buf][e]);
            async_g2l_b32(hb + c * NPIX + mt * 16 + col, &ldsGH[wave][buf][128 + e]);
        }
    };

    issueGH(0, 0);

    v8f oacc = {0.f, 0.f, 0.f, 0.f, 0.f, 0.f, 0.f, 0.f};

    for (int mt = 0; mt < 256; ++mt) {
        const int mb = mt * 16;
        const int buf = mt & 1;
        if (mt + 1 < 256) {
            issueGH(mt + 1, buf ^ 1);
            WAIT_ASYNC(8);
        } else {
            WAIT_ASYNC(0);
        }
        const float* GH = ldsGH[wave][buf];

        v2f b0, b1;
        b0[0] = GH[(2 * kh + 0) * 16 + mi];
        b0[1] = GH[(2 * kh + 1) * 16 + mi];
        b1[0] = GH[(4 + 2 * kh + 0) * 16 + mi];
        b1[1] = GH[(4 + 2 * kh + 1) * 16 + mi];

        v8f s = {0.f, 0.f, 0.f, 0.f, 0.f, 0.f, 0.f, 0.f};
        s = __builtin_amdgcn_wmma_f32_16x16x4_f32(false, a0, false, b0, (short)0, s, false, false);
        s = __builtin_amdgcn_wmma_f32_16x16x4_f32(false, a1, false, b1, (short)0, s, false, false);

        const float cmv = colmax[b * NPIX + mb + mi];  // this lane's column
#pragma unroll
        for (int r = 0; r < 8; ++r)
            etile[wave][(r + 8 * kh) * 16 + mi] = __expf(s[r] - cmv);
        // per-wave DS ops are in-order; no cross-wave sharing -> no barrier

#pragma unroll
        for (int ks = 0; ks < 4; ++ks) {
            v2f a2, b2;
            a2[0] = etile[wave][mi * 16 + ks * 4 + 2 * kh + 0];
            a2[1] = etile[wave][mi * 16 + ks * 4 + 2 * kh + 1];
            b2[0] = (mi < C8) ? GH[128 + mi * 16 + ks * 4 + 2 * kh + 0] : 0.f;
            b2[1] = (mi < C8) ? GH[128 + mi * 16 + ks * 4 + 2 * kh + 1] : 0.f;
            oacc = __builtin_amdgcn_wmma_f32_16x16x4_f32(
                       false, a2, false, b2, (short)0, oacc, false, false);
        }
    }

    if (mi < C8) {
#pragma unroll
        for (int r = 0; r < 8; ++r)
            hatt[(b * C8 + mi) * NPIX + nb + r + 8 * kh] = oacc[r];
    }
}

// ---------------------------------------------------------------------------
// i: 1x1 conv (8->64) + bias + BN + residual.
// ---------------------------------------------------------------------------
__global__ __launch_bounds__(256) void iconv_kernel(const float* __restrict__ hatt,
                                                    const float* __restrict__ i_w,
                                                    const float* __restrict__ i_b,
                                                    const float* __restrict__ i_bw,
                                                    const float* __restrict__ i_bb,
                                                    const float* __restrict__ i_bm,
                                                    const float* __restrict__ i_bv,
                                                    const float* __restrict__ out1,
                                                    float* __restrict__ out2) {
    int t = blockIdx.x * 256 + threadIdx.x;
    int b = t >> 12;
    int n = t & 4095;
    float hv[C8];
#pragma unroll
    for (int c = 0; c < C8; ++c) hv[c] = hatt[(b * C8 + c) * NPIX + n];

    for (int cm = 0; cm < CH; ++cm) {
        float acc = i_b[cm];
#pragma unroll
        for (int c = 0; c < C8; ++c) acc = fmaf(i_w[cm * C8 + c], hv[c], acc);
        float is = i_bw[cm] * rsqrtf(i_bv[cm] + 1e-5f);
        float io = i_bb[cm] - i_bm[cm] * is;
        size_t idx = (size_t)(b * CH + cm) * NPIX + n;
        out2[idx] = acc * is + io + out1[idx];
    }
}

// ---------------------------------------------------------------------------
extern "C" void kernel_launch(void* const* d_in, const int* in_sizes, int n_in,
                              void* d_out, int out_size, void* d_ws, size_t ws_size,
                              hipStream_t stream) {
    (void)in_sizes; (void)n_in; (void)out_size; (void)ws_size;

    const float* x       = (const float*)d_in[0];
    const float* conv1_w = (const float*)d_in[1];
    const float* conv2_w = (const float*)d_in[2];
    const float* f_w  = (const float*)d_in[3];
    const float* f_b  = (const float*)d_in[4];
    const float* f_bw = (const float*)d_in[5];
    const float* f_bb = (const float*)d_in[6];
    const float* f_bm = (const float*)d_in[7];
    const float* f_bv = (const float*)d_in[8];
    const float* g_w  = (const float*)d_in[9];
    const float* g_b  = (const float*)d_in[10];
    const float* g_bw = (const float*)d_in[11];
    const float* g_bb = (const float*)d_in[12];
    const float* g_bm = (const float*)d_in[13];
    const float* g_bv = (const float*)d_in[14];
    const float* gamma = (const float*)d_in[15];
    const float* h_w  = (const float*)d_in[16];
    const float* h_b  = (const float*)d_in[17];
    const float* h_bw = (const float*)d_in[18];
    const float* h_bb = (const float*)d_in[19];
    const float* h_bm = (const float*)d_in[20];
    const float* h_bv = (const float*)d_in[21];
    const float* i_w  = (const float*)d_in[22];
    const float* i_b  = (const float*)d_in[23];
    const float* i_bw = (const float*)d_in[24];
    const float* i_bb = (const float*)d_in[25];
    const float* i_bm = (const float*)d_in[26];
    const float* i_bv = (const float*)d_in[27];

    float* ws     = (float*)d_ws;
    float* out1   = ws;                               // 1,048,576 f
    float* out2   = out1 + BATCH * CH * NPIX;         // 1,048,576 f
    float* fbuf   = out2 + BATCH * CH * NPIX;         // 131,072 f
    float* gbuf   = fbuf + BATCH * C8 * NPIX;
    float* hbuf   = gbuf + BATCH * C8 * NPIX;
    float* hatt   = hbuf + BATCH * C8 * NPIX;
    float* colmax = hatt + BATCH * C8 * NPIX;         // 16,384 f
    float* colsum = colmax + BATCH * NPIX;            // 16,384 f
    unsigned short* pw1 = (unsigned short*)(colsum + BATCH * NPIX); // 36,864 bf16
    unsigned short* pw2 = pw1 + 9 * 2 * 64 * 2 * 16;

    pack_w_kernel<<<144, 256, 0, stream>>>(conv1_w, pw1);
    pack_w_kernel<<<144, 256, 0, stream>>>(conv2_w, pw2);

    conv3x3_wmma_kernel<<<BATCH * HW, 512, 0, stream>>>(x, pw1, out1);

    proj_kernel<<<64, 256, 0, stream>>>(out1,
        f_w, f_b, f_bw, f_bb, f_bm, f_bv,
        g_w, g_b, g_bw, g_bb, g_bm, g_bv,
        gamma,
        h_w, h_b, h_bw, h_bb, h_bm, h_bv,
        fbuf, gbuf, hbuf);

    att_stats_kernel<<<256, 128, 0, stream>>>(fbuf, gbuf, colmax, colsum);
    fold_h_kernel<<<(BATCH * C8 * NPIX) / 256, 256, 0, stream>>>(hbuf, colsum);
    att_apply_kernel<<<256, 128, 0, stream>>>(fbuf, gbuf, hbuf, colmax, hatt);

    iconv_kernel<<<64, 256, 0, stream>>>(hatt, i_w, i_b, i_bw, i_bb, i_bm, i_bv,
                                         out1, out2);

    conv3x3_wmma_kernel<<<BATCH * HW, 512, 0, stream>>>(out2, pw2, (float*)d_out);
}
